// AdaProjLoss_64922725647091
// MI455X (gfx1250) — matrix operand order repaired and checked by reading, not compiled, not run on hardware
//
#include <hip/hip_runtime.h>
#include <hip/hip_bf16.h>
#include <math.h>

#define IN_F   512
#define OUT_F  1000
#define BATCH  512
#define HALF_K 256   // IN_F/2

typedef _Float16 v16h __attribute__((ext_vector_type(16)));
typedef _Float16 v8h  __attribute__((ext_vector_type(8)));
typedef _Float16 v2h  __attribute__((ext_vector_type(2)));
typedef float    v8f  __attribute__((ext_vector_type(8)));

// ---------------------------------------------------------------------------
// Kernel 1: L2-normalize embeddings rows, emit f16 [BATCH, IN_F]
// ---------------------------------------------------------------------------
__global__ void __launch_bounds__(256) normalize_f16_kernel(
    const float* __restrict__ emb, _Float16* __restrict__ embH) {
  const int b = blockIdx.x;
  const float* row = emb + (size_t)b * IN_F;
  __shared__ float red[256];
  float ss = 0.f;
  for (int i = threadIdx.x; i < IN_F; i += 256) { float v = row[i]; ss += v * v; }
  red[threadIdx.x] = ss;
  __syncthreads();
  for (int off = 128; off > 0; off >>= 1) {
    if (threadIdx.x < off) red[threadIdx.x] += red[threadIdx.x + off];
    __syncthreads();
  }
  const float inv = 1.0f / fmaxf(sqrtf(red[0]), 1e-12f);
  for (int i = threadIdx.x; i < IN_F; i += 256)
    embH[(size_t)b * IN_F + i] = (_Float16)(row[i] * inv);
}

// ---------------------------------------------------------------------------
// Kernel 2: per-class GEMM + per-row L2 norm via v_wmma_f32_16x16x32_f16.
// grid = (OUT_F, 2); block = 512 threads (16 waves). Wave w owns batch rows
// [g*256 + w*16, +16). A fragments live in registers (128 VGPRs). B strips
// are double-buffered through LDS in fragment-swizzled f16 layout, with a
// register prefetch of the next tile overlapping the current WMMA chain.
// ---------------------------------------------------------------------------
__global__ void __launch_bounds__(512) proj_logits_kernel(
    const _Float16* __restrict__ embH,
    const float* __restrict__ proj,     // [OUT_F, IN_F, HALF_K]
    float* __restrict__ logits)         // [BATCH, OUT_F]
{
  const int o    = blockIdx.x;
  const int g    = blockIdx.y;          // 0..1
  const int tid  = threadIdx.x;
  const int lane = tid & 31;
  const int wave = tid >> 5;            // 0..15
  const int b0   = g * 256 + wave * 16;
  const int row  = b0 + (lane & 15);

  // double-buffered: 2 x (16 K-steps x 32 lanes x 16 halves) = 2 x 16 KB
  __shared__ __align__(32) _Float16 ldsB[2][16 * 512];

  // --- per-thread staging geometry (loop-invariant) ---
  // thread owns column kcol and row pairs i = 2p + 64j (j = 0..7).
  // K_local r = (2p)&31 is even -> halves (r, r+1) share one LDS dword.
  const int kcol = tid & 15;            // 0..15
  const int p    = tid >> 4;            // 0..31
  const int i0   = 2 * p;
  const int r    = i0 & 31;             // even
  const int s0   = p >> 4;              // 0 or 1 (K-step of first pair)
  const int doff = (kcol + ((r & 8) << 1)) * 16 + (r & 7) + ((r & 16) >> 1);  // halves offset in step
  const float* gp = proj + (size_t)o * (IN_F * HALF_K)
                         + (size_t)i0 * HALF_K + kcol;

  // --- preload A fragments: 16-bit A-matrix 16x32 layout ---
  // lanes 0-15  hold K = {0..7, 16..23}; lanes 16-31 hold K = {8..15, 24..31}
  v16h a[16];
  const int koff = (lane < 16) ? 0 : 8;
  #pragma unroll
  for (int s = 0; s < 16; ++s) {
    const _Float16* ap = embH + (size_t)row * IN_F + s * 32 + koff;
    v8h lo = *(const v8h*)(ap);         // 16B aligned
    v8h hi = *(const v8h*)(ap + 16);
    a[s] = __builtin_shufflevector(lo, hi, 0,1,2,3,4,5,6,7,8,9,10,11,12,13,14,15);
  }

  // --- prefetch tile kt=0 into registers ---
  float pf[16];
  {
    const float* q = gp;
    #pragma unroll
    for (int j = 0; j < 8; ++j) {
      pf[2*j]   = q[0];
      pf[2*j+1] = q[HALF_K];
      q += 64 * HALF_K;
    }
  }

  v8f ssq = {};
  for (int kt = 0; kt < 16; ++kt) {
    _Float16* buf = ldsB[kt & 1];
    // commit prefetched tile to LDS (8 packed dword stores)
    #pragma unroll
    for (int j = 0; j < 8; ++j) {
      v2h pk = { (_Float16)pf[2*j], (_Float16)pf[2*j+1] };
      *(v2h*)&buf[(s0 + 2*j) * 512 + doff] = pk;
    }
    __syncthreads();

    // issue next tile's global loads; latency hidden under the WMMA chain
    if (kt < 15) {
      const float* q = gp + (kt + 1) * 16;
      #pragma unroll
      for (int j = 0; j < 8; ++j) {
        pf[2*j]   = q[0];
        pf[2*j+1] = q[HALF_K];
        q += 64 * HALF_K;
      }
    }

    // --- 16 WMMAs: full K=512 reduction for this 16x16 C tile ---
    v8f c = {};
    #pragma unroll
    for (int s = 0; s < 16; ++s) {
      v16h bf = *(const v16h*)&buf[s * 512 + lane * 16];   // 32B aligned
      c = __builtin_amdgcn_wmma_f32_16x16x32_f16(
              false, a[s], false, bf, (short)0, c, false, false);
    }
    #pragma unroll
    for (int j = 0; j < 8; ++j) ssq[j] += c[j] * c[j];
  }

  // --- reduce sum-of-squares over the 16 columns (lanes within each half) ---
  #pragma unroll
  for (int j = 0; j < 8; ++j) {
    float v = ssq[j];
    v += __shfl_xor(v, 1);
    v += __shfl_xor(v, 2);
    v += __shfl_xor(v, 4);
    v += __shfl_xor(v, 8);
    ssq[j] = v;
  }
  if ((lane & 15) == 0) {
    // C/D layout: VGPR j holds row j (lanes 0-15) and row j+8 (lanes 16-31)
    const int rbase = b0 + ((lane >= 16) ? 8 : 0);
    #pragma unroll
    for (int j = 0; j < 8; ++j)
      logits[(size_t)(rbase + j) * OUT_F + o] = sqrtf(ssq[j]);
  }
}

// ---------------------------------------------------------------------------
// Kernel 3: adaptive scale s = log(OUT_F-1) / cos(mean(acos(clip(target))))
// ---------------------------------------------------------------------------
__global__ void __launch_bounds__(512) compute_s_kernel(
    const float* __restrict__ logits, const int* __restrict__ labels,
    float* __restrict__ sOut) {
  __shared__ float red[512];
  const int b = threadIdx.x;            // 512 threads == BATCH
  float t = logits[(size_t)b * OUT_F + labels[b]];
  t = fminf(fmaxf(t, -1.0f + 1e-7f), 1.0f - 1e-7f);
  red[b] = acosf(t);
  __syncthreads();
  for (int off = 256; off > 0; off >>= 1) {
    if (b < off) red[b] += red[b + off];
    __syncthreads();
  }
  if (b == 0) {
    const float meanTheta = red[0] / (float)BATCH;
    sOut[0] = logf((float)(OUT_F - 1)) / cosf(meanTheta);
  }
}

// ---------------------------------------------------------------------------
// Kernel 4: per-row log-softmax; scale logits in place; per-row NLL to ws
// ---------------------------------------------------------------------------
__global__ void __launch_bounds__(256) softmax_loss_kernel(
    float* __restrict__ logits, const int* __restrict__ labels,
    const float* __restrict__ sPtr, float* __restrict__ losses) {
  const int b = blockIdx.x;
  const int tid = threadIdx.x;
  const float s = sPtr[0];
  float* row = logits + (size_t)b * OUT_F;
  const float rawT = row[labels[b]];    // read before any in-place writes
  __shared__ float red[256];

  float m = -3.402823e38f;
  for (int o = tid; o < OUT_F; o += 256) m = fmaxf(m, row[o] * s);
  red[tid] = m;
  __syncthreads();
  for (int off = 128; off > 0; off >>= 1) {
    if (tid < off) red[tid] = fmaxf(red[tid], red[tid + off]);
    __syncthreads();
  }
  m = red[0];
  __syncthreads();

  float se = 0.f;
  for (int o = tid; o < OUT_F; o += 256) {
    const float x = row[o] * s;
    se += expf(x - m);
    row[o] = x;                         // emit scaled logits (reference output)
  }
  red[tid] = se;
  __syncthreads();
  for (int off = 128; off > 0; off >>= 1) {
    if (tid < off) red[tid] += red[tid + off];
    __syncthreads();
  }
  if (tid == 0) {
    const float lse = m + logf(red[0]);
    losses[b] = -(rawT * s - lse);
  }
}

// ---------------------------------------------------------------------------
// Kernel 5: deterministic fixed-order loss reduction
// ---------------------------------------------------------------------------
__global__ void __launch_bounds__(512) finalize_loss_kernel(
    const float* __restrict__ losses, float* __restrict__ lossOut) {
  __shared__ float red[512];
  const int t = threadIdx.x;
  red[t] = losses[t];
  __syncthreads();
  for (int off = 256; off > 0; off >>= 1) {
    if (t < off) red[t] += red[t + off];
    __syncthreads();
  }
  if (t == 0) lossOut[0] = red[0] / (float)BATCH;
}

// ---------------------------------------------------------------------------
extern "C" void kernel_launch(void* const* d_in, const int* in_sizes, int n_in,
                              void* d_out, int out_size, void* d_ws, size_t ws_size,
                              hipStream_t stream) {
  const float* emb    = (const float*)d_in[0];   // [512, 512]
  const float* proj   = (const float*)d_in[1];   // [1000, 512, 256]
  const int*   labels = (const int*)d_in[2];     // [512]

  float* logits = (float*)d_out;                        // [512*1000]
  float* loss   = logits + (size_t)BATCH * OUT_F;       // [1]

  _Float16* embH  = (_Float16*)d_ws;                                   // 512 KB
  float*    sPtr  = (float*)((char*)d_ws + (size_t)BATCH * IN_F * 2);  // 4 B
  float*    lossv = sPtr + 1;                                          // 2 KB

  normalize_f16_kernel<<<BATCH, 256, 0, stream>>>(emb, embH);

  dim3 grid(OUT_F, 2);
  proj_logits_kernel<<<grid, 512, 0, stream>>>(embH, proj, logits);

  compute_s_kernel<<<1, 512, 0, stream>>>(logits, labels, sPtr);
  softmax_loss_kernel<<<BATCH, 256, 0, stream>>>(logits, labels, sPtr, lossv);
  finalize_loss_kernel<<<1, 512, 0, stream>>>(lossv, loss);
}
